// WSDModel_39470749450485
// MI455X (gfx1250) — matrix operand order repaired and checked by reading, not compiled, not run on hardware
//
#include <hip/hip_runtime.h>
#include <hip/hip_bf16.h>
#include <cstdint>
#include <cstddef>

// ---------------------------------------------------------------------------
// WSD model pipeline for MI455X (gfx1250, wave32, WMMA).
//   embed gather -> [GEMM x@Wih^T (bf16 WMMA)] -> 100x lstm_step (h@Whh^T WMMA
//   + fused gate nonlinearity) -> layer 2 -> gather -> ctx GEMM + cls dots.
// Hidden state is double-buffered in fp32 (masked carry) AND bf16 (WMMA A
// operand) so the serial recurrence loop has zero convert ops in its K loop.
// ---------------------------------------------------------------------------

typedef __attribute__((ext_vector_type(16))) __bf16 v16bf;
typedef __attribute__((ext_vector_type(8)))  __bf16 v8bf;
typedef __attribute__((ext_vector_type(8)))  float  v8f;

namespace {
constexpr int NB  = 64;    // batch
constexpr int NT  = 100;   // time steps
constexpr int NKS = 2048;  // K selections
constexpr int NE  = 300;   // EMB
constexpr int NEP = 320;   // EMB padded to mult of 32
constexpr int NH  = 512;   // hidden
constexpr int NG  = 2048;  // 4*H gates
constexpr int NSY = 16;    // synsets per expert
}

__device__ __forceinline__ v8f wmma_bf16(v16bf a, v16bf b, v8f c) {
  return __builtin_amdgcn_wmma_f32_16x16x32_bf16(false, a, false, b, (short)0, c,
                                                 false, false);
}

// A fragment (16x32 bf16, M x K): lane L<16 holds row M=L, K {0..15};
// lane L>=16 holds row M=L-16, K {16..31}. Contiguous 32B load.
__device__ __forceinline__ v16bf load_frag_a(const __bf16* __restrict__ A, int lda,
                                             int row0, int kbase, int lane) {
  int m  = row0 + (lane & 15);
  int ko = kbase + ((lane >> 4) << 4);
  const __bf16* p = A + (size_t)m * lda + ko;
  return *(const v16bf*)(p);
}

// B fragment (32x16 bf16, K x N) for A @ W^T with W row-major [N,K]:
// lane L holds column N=L%16 = W row n, contiguous K chunk of 16 (32B load).
__device__ __forceinline__ v16bf load_frag_b(const __bf16* __restrict__ W, int ldb,
                                             int n0, int kbase, int lane) {
  int n  = n0 + (lane & 15);
  int ko = kbase + ((lane >> 4) << 4);
  return *(const v16bf*)(W + (size_t)n * ldb + ko);
}

// ---------------------------------------------------------------------------
// Utility kernels
// ---------------------------------------------------------------------------
__global__ void fill_u32(unsigned* __restrict__ p, int n) {
  int i = blockIdx.x * 256 + threadIdx.x;
  if (i < n) p[i] = 0u;
}

__global__ void cvt_bf16_pad(const float* __restrict__ src, __bf16* __restrict__ dst,
                             int rows, int cols, int dstRows, int dstCols) {
  int i = blockIdx.x * 256 + threadIdx.x;
  if (i >= dstRows * dstCols) return;
  int r = i / dstCols, c = i % dstCols;
  float v = (r < rows && c < cols) ? src[(size_t)r * cols + c] : 0.f;
  dst[i] = (__bf16)v;
}

__global__ void embed_gather(const int* __restrict__ X, const float* __restrict__ emb,
                             __bf16* __restrict__ x0) {
  int i = blockIdx.x * 256 + threadIdx.x;
  if (i >= NB * NT * NEP) return;
  int row = i / NEP, c = i % NEP;
  float v = (c < NE) ? emb[(size_t)X[row] * NE + c] : 0.f;
  x0[i] = (__bf16)v;
}

__global__ void gather_xsel(const float* __restrict__ hseq1, const int* __restrict__ sb,
                            const int* __restrict__ st, __bf16* __restrict__ xs) {
  int i = blockIdx.x * 256 + threadIdx.x;
  if (i >= NKS * 1024) return;
  int k = i >> 10, d = i & 1023;
  xs[i] = (__bf16)hseq1[((size_t)sb[k] * NT + st[k]) * 1024 + d];
}

// ---------------------------------------------------------------------------
// Generic bf16 WMMA GEMM: C[M,N] = A[M,K] @ W[N,K]^T + bias0 + bias1.
// Block = 256 threads computes a 128x128 C tile; waves 4(M)x2(N), each wave
// 2x4 tiles of 16x16 -> 8 v_wmma per 32-deep K step. M, gridN*128 must be
// covered by padded operand buffers; store guarded by Nvalid.
// ---------------------------------------------------------------------------
__global__ __launch_bounds__(256) void gemm_bf16_nt(
    const __bf16* __restrict__ A, int lda,
    const __bf16* __restrict__ W, int ldb,
    float* __restrict__ C, int ldc,
    const float* __restrict__ bias0, const float* __restrict__ bias1,
    int Kdim, int Nvalid) {
  int lane = threadIdx.x & 31;
  int w    = threadIdx.x >> 5;
  int wm   = w >> 1, wn = w & 1;
  int mBase = blockIdx.y * 128 + wm * 32;
  int nBase = blockIdx.x * 128 + wn * 64;

  v8f acc[2][4] = {};

  for (int kb = 0; kb < Kdim; kb += 32) {
    v16bf a0 = load_frag_a(A, lda, mBase, kb, lane);
    v16bf a1 = load_frag_a(A, lda, mBase + 16, kb, lane);
    v16bf b0 = load_frag_b(W, ldb, nBase,      kb, lane);
    v16bf b1 = load_frag_b(W, ldb, nBase + 16, kb, lane);
    v16bf b2 = load_frag_b(W, ldb, nBase + 32, kb, lane);
    v16bf b3 = load_frag_b(W, ldb, nBase + 48, kb, lane);
    if (kb + 32 < Kdim) {  // gfx1250 global_prefetch_b8 on next K slice
      __builtin_prefetch(A + (size_t)(mBase + (lane & 15)) * lda + kb + 32, 0, 3);
      __builtin_prefetch(W + (size_t)(nBase + (lane & 15)) * ldb + kb + 32, 0, 3);
    }
    acc[0][0] = wmma_bf16(a0, b0, acc[0][0]);
    acc[0][1] = wmma_bf16(a0, b1, acc[0][1]);
    acc[0][2] = wmma_bf16(a0, b2, acc[0][2]);
    acc[0][3] = wmma_bf16(a0, b3, acc[0][3]);
    acc[1][0] = wmma_bf16(a1, b0, acc[1][0]);
    acc[1][1] = wmma_bf16(a1, b1, acc[1][1]);
    acc[1][2] = wmma_bf16(a1, b2, acc[1][2]);
    acc[1][3] = wmma_bf16(a1, b3, acc[1][3]);
  }

  int colBase = nBase + (lane & 15);
  int rsel = (lane >> 4) << 3;  // C/D layout: vgpr i -> row i + 8*(lane/16)
#pragma unroll
  for (int j = 0; j < 4; ++j) {
    int col = colBase + j * 16;
    if (col < Nvalid) {
      float bv = 0.f;
      if (bias0) bv += bias0[col];
      if (bias1) bv += bias1[col];
#pragma unroll
      for (int i2 = 0; i2 < 2; ++i2) {
        int row = mBase + i2 * 16 + rsel;
#pragma unroll
        for (int i = 0; i < 8; ++i)
          C[(size_t)(row + i) * ldc + col] = acc[i2][j][i] + bv;
      }
    }
  }
}

// ---------------------------------------------------------------------------
// One LSTM time step, both directions (blockIdx.z), fused gates.
// gates = pre[:,t,:] + h_in @ Whh^T ; each block owns a 32-wide H-slab and
// computes its i/f/g/o columns (scattered N tiles g*512 + slab), stages the
// 64x128 gate block in LDS, then applies the nonlinearity + length mask.
// h is double-buffered across step launches; the bf16 mirror of h feeds the
// WMMA A fragments so the serial K loop has no convert ops.
// ---------------------------------------------------------------------------
__global__ __launch_bounds__(256) void lstm_step(
    const float* __restrict__ pre_f, const float* __restrict__ pre_b,
    const __bf16* __restrict__ whh_f, const __bf16* __restrict__ whh_b,
    const float* __restrict__ hf_in, float* __restrict__ hf_out,
    const __bf16* __restrict__ hfb_in, __bf16* __restrict__ hfb_out,
    float* __restrict__ cf,
    const float* __restrict__ hb_in, float* __restrict__ hb_out,
    const __bf16* __restrict__ hbb_in, __bf16* __restrict__ hbb_out,
    float* __restrict__ cb,
    float* __restrict__ hseq, const int* __restrict__ lengths, int t_step) {
  int dir = blockIdx.z;
  int hsb = blockIdx.x * 32;  // H-slab base
  const float*  pre   = dir ? pre_b : pre_f;
  const __bf16* whh   = dir ? whh_b : whh_f;
  const float*  hin   = dir ? hb_in : hf_in;
  float*        hout  = dir ? hb_out : hf_out;
  const __bf16* hbin  = dir ? hbb_in : hfb_in;
  __bf16*       hbout = dir ? hbb_out : hfb_out;
  float*        cst   = dir ? cb : cf;
  int t_eff = dir ? (NT - 1 - t_step) : t_step;

  __shared__ float sg[64 * 128];  // 64 rows x (4 gates x 32 cols)

  int lane = threadIdx.x & 31;
  int w    = threadIdx.x >> 5;
  int wm   = w >> 2;      // 0..1 -> M base wm*32
  int wn   = w & 3;       // 0..3 -> N tile pair j0 = 2*wn (8 tiles total)
  int mBase = wm * 32;
  int j0 = wn * 2;
  int nA = ((j0    ) >> 1) * NH + hsb + ((j0    ) & 1) * 16;
  int nB = ((j0 + 1) >> 1) * NH + hsb + ((j0 + 1) & 1) * 16;

  v8f acc[2][2] = {};
  for (int kb = 0; kb < NH; kb += 32) {
    v16bf a0 = load_frag_a(hbin, NH, mBase, kb, lane);
    v16bf a1 = load_frag_a(hbin, NH, mBase + 16, kb, lane);
    v16bf b0 = load_frag_b(whh, NH, nA, kb, lane);
    v16bf b1 = load_frag_b(whh, NH, nB, kb, lane);
    if (kb + 32 < NH) {
      __builtin_prefetch(whh + (size_t)(nA + (lane & 15)) * NH + kb + 32, 0, 3);
      __builtin_prefetch(whh + (size_t)(nB + (lane & 15)) * NH + kb + 32, 0, 3);
    }
    acc[0][0] = wmma_bf16(a0, b0, acc[0][0]);
    acc[0][1] = wmma_bf16(a0, b1, acc[0][1]);
    acc[1][0] = wmma_bf16(a1, b0, acc[1][0]);
    acc[1][1] = wmma_bf16(a1, b1, acc[1][1]);
  }

  int rsel = (lane >> 4) << 3;
#pragma unroll
  for (int jj = 0; jj < 2; ++jj) {
    int j = j0 + jj;
    int cloc = (j >> 1) * 32 + (j & 1) * 16 + (lane & 15);
#pragma unroll
    for (int i2 = 0; i2 < 2; ++i2) {
      int row = mBase + i2 * 16 + rsel;
#pragma unroll
      for (int i = 0; i < 8; ++i) sg[(row + i) * 128 + cloc] = acc[i2][jj][i];
    }
  }
  __syncthreads();

  for (int r = threadIdx.x; r < 64 * 32; r += 256) {
    int m = r >> 5, hl = r & 31;
    int hcol = hsb + hl;
    size_t prow = ((size_t)m * NT + t_eff) * NG;
    float gi = sg[m * 128 +  0 + hl] + pre[prow + 0 * NH + hcol];
    float gf = sg[m * 128 + 32 + hl] + pre[prow + 1 * NH + hcol];
    float gg = sg[m * 128 + 64 + hl] + pre[prow + 2 * NH + hcol];
    float go = sg[m * 128 + 96 + hl] + pre[prow + 3 * NH + hcol];
    float si = 1.f / (1.f + __expf(-gi));
    float sf = 1.f / (1.f + __expf(-gf));
    float so = 1.f / (1.f + __expf(-go));
    float tg = tanhf(gg);
    int idx = m * NH + hcol;
    float c_old = cst[idx];
    float h_old = hin[idx];
    float c_new = sf * c_old + si * tg;
    float h_new = so * tanhf(c_new);
    bool valid = t_eff < lengths[m];
    float cw = valid ? c_new : c_old;
    float hw = valid ? h_new : h_old;
    cst[idx] = cw;
    hout[idx] = hw;
    hbout[idx] = (__bf16)hw;  // bf16 mirror feeds next step's WMMA A operand
    hseq[((size_t)m * NT + t_eff) * 1024 + dir * NH + hcol] = hw;
  }
}

// ---------------------------------------------------------------------------
// cls[k,s] = dot(exp_W[e[k],s,:], xsel[k]) + exp_b[e[k],s]  (bandwidth-bound)
// One block per k, one wave per 2 synsets, float4 loads + wave32 reduce.
// ---------------------------------------------------------------------------
__global__ __launch_bounds__(256) void cls_kernel(
    const float* __restrict__ hseq1, const int* __restrict__ sb,
    const int* __restrict__ st, const int* __restrict__ eidx,
    const float* __restrict__ expW, const float* __restrict__ expb,
    float* __restrict__ out) {
  int k = blockIdx.x;
  int w = threadIdx.x >> 5;
  int lane = threadIdx.x & 31;
  int b = sb[k], t = st[k], e = eidx[k];
  const float* x = hseq1 + ((size_t)b * NT + t) * 1024;
  for (int s = w; s < NSY; s += 8) {
    const float* wr = expW + ((size_t)e * NSY + s) * 1024;
    float acc = 0.f;
    for (int d = lane * 4; d < 1024; d += 128) {
      float4 xv = *(const float4*)(x + d);
      float4 wv = *(const float4*)(wr + d);
      acc += xv.x * wv.x + xv.y * wv.y + xv.z * wv.z + xv.w * wv.w;
    }
    for (int off = 16; off; off >>= 1) acc += __shfl_down(acc, off, 32);
    if (lane == 0) out[k * NSY + s] = acc + expb[e * NSY + s];
  }
}

// ---------------------------------------------------------------------------
extern "C" void kernel_launch(void* const* d_in, const int* in_sizes, int n_in,
                              void* d_out, int out_size, void* d_ws, size_t ws_size,
                              hipStream_t stream) {
  (void)in_sizes; (void)n_in; (void)out_size; (void)ws_size;
  const int*   X    = (const int*)d_in[0];
  const int*   Xlen = (const int*)d_in[1];
  const int*   selb = (const int*)d_in[2];
  const int*   selt = (const int*)d_in[3];
  const int*   eidx = (const int*)d_in[4];
  const float* emb  = (const float*)d_in[5];
  // order: l0f, l0b, l1f, l1b
  const float* wih[4] = {(const float*)d_in[6],  (const float*)d_in[10],
                         (const float*)d_in[14], (const float*)d_in[18]};
  const float* whh[4] = {(const float*)d_in[7],  (const float*)d_in[11],
                         (const float*)d_in[15], (const float*)d_in[19]};
  const float* bih[4] = {(const float*)d_in[8],  (const float*)d_in[12],
                         (const float*)d_in[16], (const float*)d_in[20]};
  const float* bhh[4] = {(const float*)d_in[9],  (const float*)d_in[13],
                         (const float*)d_in[17], (const float*)d_in[21]};
  const float* outW = (const float*)d_in[22];
  const float* outb = (const float*)d_in[23];
  const float* expW = (const float*)d_in[24];
  const float* expb = (const float*)d_in[25];

  char* p = (char*)d_ws;
  auto carve = [&](size_t bytes) -> char* {
    char* r = p;
    p += (bytes + 255) & ~(size_t)255;
    return r;
  };
  const size_t MBT = (size_t)NB * NT;  // 6400
  __bf16* x0bf    = (__bf16*)carve(MBT * NEP * 2);
  __bf16* wihbf0f = (__bf16*)carve((size_t)NG * NEP * 2);
  __bf16* wihbf0b = (__bf16*)carve((size_t)NG * NEP * 2);
  __bf16* wihbf1f = (__bf16*)carve((size_t)NG * 1024 * 2);
  __bf16* wihbf1b = (__bf16*)carve((size_t)NG * 1024 * 2);
  __bf16* whhbf[4];
  for (int i = 0; i < 4; ++i) whhbf[i] = (__bf16*)carve((size_t)NG * NH * 2);
  float* pre_f   = (float*)carve(MBT * NG * 4);
  float* pre_b   = (float*)carve(MBT * NG * 4);
  float* hseq0   = (float*)carve(MBT * 1024 * 4);
  __bf16* hseq0bf= (__bf16*)carve(MBT * 1024 * 2);
  float* hseq1   = (float*)carve(MBT * 1024 * 4);
  float* states  = (float*)carve((size_t)6 * NB * NH * 4);  // hf0,hf1,hb0,hb1,cf,cb
  __bf16* statesb= (__bf16*)carve((size_t)4 * NB * NH * 2); // bf16 mirrors of h
  float* hf[2] = {states, states + NB * NH};
  float* hb[2] = {states + 2 * NB * NH, states + 3 * NB * NH};
  float* cf = states + 4 * NB * NH;
  float* cb = states + 5 * NB * NH;
  __bf16* hfb[2] = {statesb, statesb + NB * NH};
  __bf16* hbb[2] = {statesb + 2 * NB * NH, statesb + 3 * NB * NH};
  __bf16* outWbf = (__bf16*)carve((size_t)384 * 1024 * 2);
  __bf16* xselbf = (__bf16*)carve((size_t)NKS * 1024 * 2);

  auto g1 = [](size_t n) { return dim3((unsigned)((n + 255) / 256)); };
  const int stateWords = (6 * NB * NH * 4 + 4 * NB * NH * 2) / 4;  // fp32+bf16 states

  // Weight conversions fp32 -> bf16 (pad K of l0 w_ih to 320, out_W rows to 384).
  cvt_bf16_pad<<<g1((size_t)NG * NEP), 256, 0, stream>>>(wih[0], wihbf0f, NG, NE, NG, NEP);
  cvt_bf16_pad<<<g1((size_t)NG * NEP), 256, 0, stream>>>(wih[1], wihbf0b, NG, NE, NG, NEP);
  cvt_bf16_pad<<<g1((size_t)NG * 1024), 256, 0, stream>>>(wih[2], wihbf1f, NG, 1024, NG, 1024);
  cvt_bf16_pad<<<g1((size_t)NG * 1024), 256, 0, stream>>>(wih[3], wihbf1b, NG, 1024, NG, 1024);
  for (int i = 0; i < 4; ++i)
    cvt_bf16_pad<<<g1((size_t)NG * NH), 256, 0, stream>>>(whh[i], whhbf[i], NG, NH, NG, NH);
  cvt_bf16_pad<<<g1((size_t)384 * 1024), 256, 0, stream>>>(outW, outWbf, NE, 1024, 384, 1024);

  // Embedding gather (+pad, +bf16).
  embed_gather<<<g1(MBT * NEP), 256, 0, stream>>>(X, emb, x0bf);

  // ---- Layer 0: hoisted input projection, then 100 recurrence steps ----
  gemm_bf16_nt<<<dim3(16, 50), 256, 0, stream>>>(x0bf, NEP, wihbf0f, NEP, pre_f, NG,
                                                 bih[0], bhh[0], NEP, NG);
  gemm_bf16_nt<<<dim3(16, 50), 256, 0, stream>>>(x0bf, NEP, wihbf0b, NEP, pre_b, NG,
                                                 bih[1], bhh[1], NEP, NG);
  fill_u32<<<g1(stateWords), 256, 0, stream>>>((unsigned*)states, stateWords);
  for (int t = 0; t < NT; ++t) {
    int a = t & 1, o = 1 - a;
    lstm_step<<<dim3(16, 1, 2), 256, 0, stream>>>(
        pre_f, pre_b, whhbf[0], whhbf[1],
        hf[a], hf[o], hfb[a], hfb[o], cf,
        hb[a], hb[o], hbb[a], hbb[o], cb,
        hseq0, Xlen, t);
  }

  // ---- Layer 1 ----
  cvt_bf16_pad<<<g1(MBT * 1024), 256, 0, stream>>>(hseq0, hseq0bf, (int)MBT, 1024,
                                                   (int)MBT, 1024);
  gemm_bf16_nt<<<dim3(16, 50), 256, 0, stream>>>(hseq0bf, 1024, wihbf1f, 1024, pre_f, NG,
                                                 bih[2], bhh[2], 1024, NG);
  gemm_bf16_nt<<<dim3(16, 50), 256, 0, stream>>>(hseq0bf, 1024, wihbf1b, 1024, pre_b, NG,
                                                 bih[3], bhh[3], 1024, NG);
  fill_u32<<<g1(stateWords), 256, 0, stream>>>((unsigned*)states, stateWords);
  for (int t = 0; t < NT; ++t) {
    int a = t & 1, o = 1 - a;
    lstm_step<<<dim3(16, 1, 2), 256, 0, stream>>>(
        pre_f, pre_b, whhbf[2], whhbf[3],
        hf[a], hf[o], hfb[a], hfb[o], cf,
        hb[a], hb[o], hbb[a], hbb[o], cb,
        hseq1, Xlen, t);
  }

  // ---- Heads ----
  gather_xsel<<<g1((size_t)NKS * 1024), 256, 0, stream>>>(hseq1, selb, selt, xselbf);
  // ctx = xsel @ out_W^T + out_b  (M=2048, N=300 valid of 384, K=1024)
  gemm_bf16_nt<<<dim3(3, 16), 256, 0, stream>>>(xselbf, 1024, outWbf, 1024,
                                                (float*)d_out, NE, outb, nullptr,
                                                1024, NE);
  cls_kernel<<<dim3(NKS), 256, 0, stream>>>(hseq1, selb, selt, eidx, expW, expb,
                                            (float*)d_out + (size_t)NKS * NE);
}
